// myMultiheadAttention_88021059764812
// MI455X (gfx1250) — compile-verified, matrix operand order checked by reading, not run on hardware
//
#include <hip/hip_runtime.h>

// ---------------- types ----------------
typedef __attribute__((ext_vector_type(16))) __bf16 v16bf;
typedef __attribute__((ext_vector_type(8)))  __bf16 v8bf;
typedef __attribute__((ext_vector_type(4)))  __bf16 v4bf;
typedef __attribute__((ext_vector_type(8)))  float  v8f;
typedef __attribute__((ext_vector_type(4)))  float  v4f;

// ---------------- problem dims ----------------
constexpr int S_LEN = 2048;
constexpr int BATCH = 2;
constexpr int EMB   = 1024;
constexpr int NH    = 16;
constexpr int HD    = 64;              // EMB / NH
constexpr int NROWS = S_LEN * BATCH;   // 4096 rows of the (S,B,*) tensors

// ---------------- WMMA helpers ----------------
// D = A(16x32 bf16) * B(32x16 bf16) + C(16x16 f32)
__device__ __forceinline__ v8f wmma_bf16(v16bf a, v16bf b, v8f c) {
  return __builtin_amdgcn_wmma_f32_16x16x32_bf16(
      /*neg_a=*/false, a, /*neg_b=*/false, b,
      /*c_mod=*/(short)0, c, /*reuse_a=*/false, /*reuse_b=*/false);
}

// Load one 16-bit A- or B-fragment from a row-major matrix.
// ISA 7.12.2 layout: lane half h, VGPR v -> K = (v/4)*16 + h*8 + (v%4)*2 (+pair)
// => per lane: 8 contiguous bf16 at k0 + h*8 and 8 more at k0 + 16 + h*8.
__device__ __forceinline__ v16bf load_frag_bf16(const __bf16* __restrict__ rowptr,
                                                int k0, int half) {
  v8bf lo = *(const v8bf*)(rowptr + k0 + half * 8);
  v8bf hi = *(const v8bf*)(rowptr + k0 + 16 + half * 8);
  union { v16bf v; v8bf h[2]; } u;
  u.h[0] = lo; u.h[1] = hi;
  return u.v;
}

__device__ __forceinline__ v8f zero_v8f() {
  v8f z;
#pragma unroll
  for (int e = 0; e < 8; ++e) z[e] = 0.0f;
  return z;
}

// ---------------- kernel 0: f32 -> bf16 convert ----------------
__global__ void k_cvt(const float* __restrict__ in, __bf16* __restrict__ out, int n4) {
  int i = blockIdx.x * blockDim.x + threadIdx.x;
  if (i < n4) {
    v4f x = ((const v4f*)in)[i];
    v4bf y;
#pragma unroll
    for (int e = 0; e < 4; ++e) y[e] = (__bf16)x[e];
    ((v4bf*)out)[i] = y;
  }
}

// ---------------- kernel 1: fused QKV projection ----------------
// C[r,j] = X[r,:] . W[z*E + j, :] + bias[z*E + j]
// z=0 -> Qb[(b*16+h)][s][hd] *= 1/8 ; z=1 -> Kb same layout ; z=2 -> Vt[(b*16+h)][hd][s]
__global__ void __launch_bounds__(128)
k_qkv_gemm(const __bf16* __restrict__ Xq, const __bf16* __restrict__ Xk,
           const __bf16* __restrict__ Xv, const __bf16* __restrict__ W,
           const float* __restrict__ bias,
           __bf16* __restrict__ Qb, __bf16* __restrict__ Kb, __bf16* __restrict__ Vt) {
  const int lane = threadIdx.x & 31;
  const int wave = threadIdx.x >> 5;       // 0..3
  const int half = lane >> 4, l16 = lane & 15;
  const int which = blockIdx.z;            // 0=q 1=k 2=v
  const __bf16* X = (which == 0) ? Xq : (which == 1 ? Xk : Xv);
  const int r0 = blockIdx.x * 64;
  const int j0 = blockIdx.y * 128 + wave * 32;

  v8f acc[4][2];
#pragma unroll
  for (int mf = 0; mf < 4; ++mf)
#pragma unroll
    for (int nf = 0; nf < 2; ++nf) acc[mf][nf] = zero_v8f();

  for (int k0 = 0; k0 < EMB; k0 += 32) {
    v16bf afr[4], bfr[2];
#pragma unroll
    for (int mf = 0; mf < 4; ++mf)
      afr[mf] = load_frag_bf16(X + (size_t)(r0 + mf * 16 + l16) * EMB, k0, half);
#pragma unroll
    for (int nf = 0; nf < 2; ++nf)
      bfr[nf] = load_frag_bf16(W + (size_t)(which * EMB + j0 + nf * 16 + l16) * EMB, k0, half);
#pragma unroll
    for (int mf = 0; mf < 4; ++mf)
#pragma unroll
      for (int nf = 0; nf < 2; ++nf)
        acc[mf][nf] = wmma_bf16(afr[mf], bfr[nf], acc[mf][nf]);
  }

#pragma unroll
  for (int mf = 0; mf < 4; ++mf)
#pragma unroll
    for (int nf = 0; nf < 2; ++nf) {
      const int j = j0 + nf * 16 + l16;
      const float bj = bias[which * EMB + j];
      const int h = j >> 6, hd = j & 63;
#pragma unroll
      for (int v = 0; v < 8; ++v) {
        const int r = r0 + mf * 16 + half * 8 + v;
        const int s = r >> 1, b = r & 1;     // BATCH == 2
        const float val = acc[mf][nf][v] + bj;
        if (which == 0)
          Qb[((size_t)(b * NH + h) * S_LEN + s) * HD + hd] = (__bf16)(val * 0.125f);
        else if (which == 1)
          Kb[((size_t)(b * NH + h) * S_LEN + s) * HD + hd] = (__bf16)val;
        else
          Vt[((size_t)(b * NH + h) * HD + hd) * S_LEN + s] = (__bf16)val;
      }
    }
}

// ---------------- kernel 2: fused attention (scores+softmax+probs+attn@V) ---
// Computes scores TRANSPOSED: S^T = K . Q^T  (per 16x16 tile: k rows in
// VGPRs/halves, q cols in lanes).  The S^T C-fragment register layout is
// exactly the A-fragment layout of P for the attn@V WMMA, so normalized
// probabilities flow from softmax registers straight into v_wmma operands.
// Phase 1: online per-lane (per-q) max/sum, no global writes.
// Phase 2: recompute tiles, write normalized probs ONCE, accumulate ctx.
__global__ void __launch_bounds__(128)
k_attention(const __bf16* __restrict__ Qb, const __bf16* __restrict__ Kb,
            const __bf16* __restrict__ Vt, const float* __restrict__ pb,
            float* __restrict__ attn, __bf16* __restrict__ Ctx) {
  const int lane = threadIdx.x & 31;
  const int wave = threadIdx.x >> 5;
  const int half = lane >> 4, l16 = lane & 15;
  const int bh = blockIdx.y;               // b*16 + h
  const int b = bh >> 4, h = bh & 15;
  const int q0 = blockIdx.x * 64 + wave * 16;

  // Resident B-fragments of Q^T: B[hd][q], row q0+l16 of Qb (Q pre-scaled)
  const __bf16* Qrow = Qb + ((size_t)bh * S_LEN + q0 + l16) * HD;
  const v16bf bq0 = load_frag_bf16(Qrow, 0, half);
  const v16bf bq1 = load_frag_bf16(Qrow, 32, half);

  const __bf16* Kbase = Kb + (size_t)bh * S_LEN * HD;
  const float* pbrow = pb + ((size_t)b * S_LEN + (q0 + l16)) * S_LEN;
  float* arow = attn + ((size_t)bh * S_LEN + (q0 + l16)) * S_LEN;

  // ---- phase 1: online max / sum (per lane covers k = half*8+v mod 16) ----
  float lm = -1.0e30f, ls = 0.0f;
  for (int kc = 0; kc < S_LEN; kc += 16) {
    const __bf16* Krow = Kbase + (size_t)(kc + l16) * HD;
    __builtin_prefetch(Kbase + (size_t)(((kc + 64) & (S_LEN - 1)) + l16) * HD, 0, 1);
    v16bf ak0 = load_frag_bf16(Krow, 0, half);
    v16bf ak1 = load_frag_bf16(Krow, 32, half);
    v8f acc = zero_v8f();
    acc = wmma_bf16(ak0, bq0, acc);
    acc = wmma_bf16(ak1, bq1, acc);
    v4f bia0 = *(const v4f*)(pbrow + kc + half * 8);
    v4f bia1 = *(const v4f*)(pbrow + kc + half * 8 + 4);
    float x[8];
    float tm = -1.0e30f;
#pragma unroll
    for (int v = 0; v < 8; ++v) {
      x[v] = acc[v] + ((v < 4) ? bia0[v] : bia1[v - 4]);
      tm = fmaxf(tm, x[v]);
    }
    const float nm = fmaxf(lm, tm);
    float s = 0.0f;
#pragma unroll
    for (int v = 0; v < 8; ++v) s += __expf(x[v] - nm);
    ls = ls * __expf(lm - nm) + s;
    lm = nm;
  }
  // combine the two halves (same q lives in lanes L and L+16)
  const float om = __shfl_xor(lm, 16, 32);
  const float M = fmaxf(lm, om);
  float sadj = ls * __expf(lm - M);
  const float total = sadj + __shfl_xor(sadj, 16, 32);
  const float invS = 1.0f / total;

  // ---- phase 2: recompute, emit probs once, fused attn@V ----
  v8f cacc[4];
#pragma unroll
  for (int nf = 0; nf < 4; ++nf) cacc[nf] = zero_v8f();

  for (int kp = 0; kp < S_LEN; kp += 32) {
    v16bf afrP;
#pragma unroll
    for (int jt = 0; jt < 2; ++jt) {
      const int kc = kp + jt * 16;
      const __bf16* Krow = Kbase + (size_t)(kc + l16) * HD;
      v16bf ak0 = load_frag_bf16(Krow, 0, half);
      v16bf ak1 = load_frag_bf16(Krow, 32, half);
      v8f acc = zero_v8f();
      acc = wmma_bf16(ak0, bq0, acc);
      acc = wmma_bf16(ak1, bq1, acc);
      v4f bia0 = *(const v4f*)(pbrow + kc + half * 8);
      v4f bia1 = *(const v4f*)(pbrow + kc + half * 8 + 4);
      v4f p0, p1;
#pragma unroll
      for (int e = 0; e < 4; ++e) p0[e] = __expf(acc[e] + bia0[e] - M) * invS;
#pragma unroll
      for (int e = 0; e < 4; ++e) p1[e] = __expf(acc[4 + e] + bia1[e] - M) * invS;
      // the single pass over the 537MB attention matrix
      *(v4f*)(arow + kc + half * 8) = p0;
      *(v4f*)(arow + kc + half * 8 + 4) = p1;
      // S^T C-frag == A-frag of P: elems jt*8 + e  <->  k = jt*16 + half*8 + e
#pragma unroll
      for (int e = 0; e < 4; ++e) {
        afrP[jt * 8 + e]     = (__bf16)p0[e];
        afrP[jt * 8 + 4 + e] = (__bf16)p1[e];
      }
    }
    v16bf bv[4];
#pragma unroll
    for (int nf = 0; nf < 4; ++nf)
      bv[nf] = load_frag_bf16(Vt + ((size_t)bh * HD + nf * 16 + l16) * S_LEN, kp, half);
#pragma unroll
    for (int nf = 0; nf < 4; ++nf) cacc[nf] = wmma_bf16(afrP, bv[nf], cacc[nf]);
  }

  // store context: rows m = half*8+v <-> q, cols = hd (coalesced bf16)
#pragma unroll
  for (int nf = 0; nf < 4; ++nf) {
    const int j = h * 64 + nf * 16 + l16;
#pragma unroll
    for (int v = 0; v < 8; ++v) {
      const int q = q0 + half * 8 + v;
      const int r = q * BATCH + b;
      Ctx[(size_t)r * EMB + j] = (__bf16)cacc[nf][v];
    }
  }
}

// ---------------- kernel 3: output projection ----------------
__global__ void __launch_bounds__(128)
k_out_proj(const __bf16* __restrict__ Ctx, const __bf16* __restrict__ Wo,
           const float* __restrict__ ob, float* __restrict__ Out) {
  const int lane = threadIdx.x & 31;
  const int wave = threadIdx.x >> 5;
  const int half = lane >> 4, l16 = lane & 15;
  const int r0 = blockIdx.x * 64;
  const int j0 = blockIdx.y * 128 + wave * 32;

  v8f acc[4][2];
#pragma unroll
  for (int mf = 0; mf < 4; ++mf)
#pragma unroll
    for (int nf = 0; nf < 2; ++nf) acc[mf][nf] = zero_v8f();

  for (int k0 = 0; k0 < EMB; k0 += 32) {
    v16bf afr[4], bfr[2];
#pragma unroll
    for (int mf = 0; mf < 4; ++mf)
      afr[mf] = load_frag_bf16(Ctx + (size_t)(r0 + mf * 16 + l16) * EMB, k0, half);
#pragma unroll
    for (int nf = 0; nf < 2; ++nf)
      bfr[nf] = load_frag_bf16(Wo + (size_t)(j0 + nf * 16 + l16) * EMB, k0, half);
#pragma unroll
    for (int mf = 0; mf < 4; ++mf)
#pragma unroll
      for (int nf = 0; nf < 2; ++nf)
        acc[mf][nf] = wmma_bf16(afr[mf], bfr[nf], acc[mf][nf]);
  }

#pragma unroll
  for (int mf = 0; mf < 4; ++mf)
#pragma unroll
    for (int nf = 0; nf < 2; ++nf) {
      const int j = j0 + nf * 16 + l16;
      const float bj = ob[j];
#pragma unroll
      for (int v = 0; v < 8; ++v) {
        const int r = r0 + mf * 16 + half * 8 + v;
        Out[(size_t)r * EMB + j] = acc[mf][nf][v] + bj;
      }
    }
}

// ---------------- host launcher ----------------
extern "C" void kernel_launch(void* const* d_in, const int* in_sizes, int n_in,
                              void* d_out, int out_size, void* d_ws, size_t ws_size,
                              hipStream_t stream) {
  const float* query = (const float*)d_in[0];
  const float* key   = (const float*)d_in[1];
  const float* value = (const float*)d_in[2];
  const float* pbias = (const float*)d_in[3];
  const float* ipw   = (const float*)d_in[4];
  const float* ipb   = (const float*)d_in[5];
  const float* outw  = (const float*)d_in[6];
  const float* outb  = (const float*)d_in[7];

  float* out  = (float*)d_out;                       // (S,B,E) f32
  float* attn = out + (size_t)NROWS * EMB;           // (B*H,S,S) f32

  // workspace layout (bf16 elements), total 64 MB
  __bf16* base = (__bf16*)d_ws;
  size_t o = 0;
  __bf16* Xq   = base + o; o += (size_t)NROWS * EMB;
  __bf16* Xk   = base + o; o += (size_t)NROWS * EMB;
  __bf16* Xv   = base + o; o += (size_t)NROWS * EMB;
  __bf16* Wqkv = base + o; o += (size_t)3 * EMB * EMB;
  __bf16* Wo   = base + o; o += (size_t)EMB * EMB;
  __bf16* Qb   = base + o; o += (size_t)BATCH * NH * S_LEN * HD;
  __bf16* Kb   = base + o; o += (size_t)BATCH * NH * S_LEN * HD;
  __bf16* Vt   = base + o; o += (size_t)BATCH * NH * S_LEN * HD;
  __bf16* Ctx  = base + o; o += (size_t)NROWS * EMB;

  auto cvt = [&](const float* in, __bf16* outp, size_t n) {
    int n4 = (int)(n / 4);
    k_cvt<<<(n4 + 255) / 256, 256, 0, stream>>>(in, outp, n4);
  };
  cvt(query, Xq, (size_t)NROWS * EMB);
  cvt(key,   Xk, (size_t)NROWS * EMB);
  cvt(value, Xv, (size_t)NROWS * EMB);
  cvt(ipw,   Wqkv, (size_t)3 * EMB * EMB);
  cvt(outw,  Wo, (size_t)EMB * EMB);

  dim3 g1(NROWS / 64, EMB / 128, 3);
  k_qkv_gemm<<<g1, 128, 0, stream>>>(Xq, Xk, Xv, Wqkv, ipb, Qb, Kb, Vt);

  dim3 g2(S_LEN / 64, BATCH * NH);
  k_attention<<<g2, 128, 0, stream>>>(Qb, Kb, Vt, pbias, attn, Ctx);

  dim3 g3(NROWS / 64, EMB / 128);
  k_out_proj<<<g3, 128, 0, stream>>>(Ctx, Wo, outb, out);
}
